// LSTMTagger_75668733821508
// MI455X (gfx1250) — matrix operand order repaired and checked by reading, not compiled, not run on hardware
//
#include <hip/hip_runtime.h>
#include <hip/hip_bf16.h>

// ---------------- types for WMMA / async copy ----------------
typedef __bf16  v16bf __attribute__((ext_vector_type(16)));
typedef float   v8f   __attribute__((ext_vector_type(8)));
typedef int     v4i   __attribute__((vector_size(16)));
typedef v4i __attribute__((address_space(1)))* gv4i_p;   // global (AS1) 16B vector ptr
typedef v4i __attribute__((address_space(3)))* lv4i_p;   // LDS (AS3) 16B vector ptr

// ---------------- problem constants ----------------
#define BATCH 64
#define SEQ   256
#define HID   1024
#define GATES 4096           // 4*HID
#define DIN   14
#define ROWS  16384          // SEQ*BATCH
#define TAGS  50
#define TAGP  64             // padded tag dim
#define LDS_ROWS 32          // h_prev rows staged per block

// ---------------- workspace layout (bytes) ----------------
#define OFF_X       ((size_t)0)                         // [16384][16] f32      = 1 MiB
#define OFF_WHHF    (OFF_X      + (size_t)ROWS*16*4)    // [4096][1024] bf16    = 8 MiB
#define OFF_WHHB    (OFF_WHHF   + (size_t)GATES*HID*2)  // 8 MiB
#define OFF_HSF     (OFF_WHHB   + (size_t)GATES*HID*2)  // [256][64][1024] bf16 = 32 MiB
#define OFF_HSB     (OFF_HSF    + (size_t)SEQ*BATCH*HID*2)
#define OFF_CF      (OFF_HSB    + (size_t)SEQ*BATCH*HID*2) // [64][1024] f32
#define OFF_CB      (OFF_CF     + (size_t)BATCH*HID*4)
#define OFF_HZERO   (OFF_CB     + (size_t)BATCH*HID*4)     // [64][1024] bf16
#define OFF_FCW     (OFF_HZERO  + (size_t)BATCH*HID*2)     // [64][2048] bf16
#define OFF_TAG     (OFF_FCW    + (size_t)TAGP*2*HID*2)    // [16384][64] f32

// ---------------- helpers ----------------
static __device__ __forceinline__ unsigned short f32_to_bf16(float f) {
  unsigned int u = __float_as_uint(f);
  unsigned int r = (u + 0x7FFFu + ((u >> 16) & 1u)) >> 16;
  return (unsigned short)r;
}
static __device__ __forceinline__ __bf16 us_as_bf16(unsigned short s) {
  union { unsigned short u; __bf16 h; } cv; cv.u = s; return cv.h;
}
static __device__ __forceinline__ float sigm(float x) {
  return 1.0f / (1.0f + expf(-x));
}
static __device__ __forceinline__ void wait_async0() {
#if __has_builtin(__builtin_amdgcn_s_wait_asynccnt)
  __builtin_amdgcn_s_wait_asynccnt(0);
#else
  asm volatile("s_wait_asynccnt 0x0" ::: "memory");
#endif
}

// ---------------- kernel 1: embeddings + char CNN -> x[16384][16] ----------------
__global__ void prep_embed(const int* __restrict__ char_idx,
                           const int* __restrict__ sent,
                           const float* __restrict__ wemb,
                           const float* __restrict__ cemb,
                           const float* __restrict__ cnnw,   // [4][18]
                           const float* __restrict__ cnnb,   // [4]
                           float* __restrict__ x) {
  int q = blockIdx.x * blockDim.x + threadIdx.x;
  if (q >= ROWS) return;
  float* xr = x + (size_t)q * 16;
  int w = sent[q];
  for (int d = 0; d < 10; ++d) xr[d] = wemb[(size_t)w * 10 + d];
  const int* ci = char_idx + (size_t)q * 16;
  for (int l = 0; l < 4; ++l) {
    float mx = -3.4e38f;
    for (int wp = 0; wp < 14; ++wp) {
      float acc = cnnb[l];
      for (int i = 0; i < 3; ++i) {
        int c = ci[wp + i];
        for (int e = 0; e < 6; ++e)
          acc += cemb[(size_t)c * 6 + e] * cnnw[l * 18 + i * 6 + e];
      }
      mx = fmaxf(mx, acc);
    }
    xr[10 + l] = mx;
  }
  xr[14] = 0.f; xr[15] = 0.f;
}

// ---------------- conversion / init kernels ----------------
__global__ void cvt_bf16(const float* __restrict__ src, unsigned short* __restrict__ dst, int n) {
  int i = blockIdx.x * blockDim.x + threadIdx.x;
  if (i < n) dst[i] = f32_to_bf16(src[i]);
}
__global__ void cvt_fcw(const float* __restrict__ fcw, unsigned short* __restrict__ dst) {
  int i = blockIdx.x * blockDim.x + threadIdx.x;   // over 64*2048
  if (i >= TAGP * 2 * HID) return;
  int t = i >> 11, k = i & 2047;
  float v = (t < TAGS) ? fcw[(size_t)t * 2 * HID + k] : 0.f;
  dst[i] = f32_to_bf16(v);
}
__global__ void zero_u32(unsigned int* __restrict__ p, int n) {
  int i = blockIdx.x * blockDim.x + threadIdx.x;
  if (i < n) p[i] = 0u;
}

// One gate: load B fragment from whh row block, do 2 WMMAs (two M-tiles).
static __device__ __forceinline__ void gate_mma(const unsigned short* __restrict__ bw,
                                                int kk, int kbaseB,
                                                const v16bf& a0, const v16bf& a1,
                                                v8f& acc0, v8f& acc1) {
  v16bf b;
  #pragma unroll
  for (int j = 0; j < 16; ++j) b[j] = us_as_bf16(bw[kk + kbaseB + j]);
  acc0 = __builtin_amdgcn_wmma_f32_16x16x32_bf16(false, a0, false, b, (short)0, acc0, false, false);
  acc1 = __builtin_amdgcn_wmma_f32_16x16x32_bf16(false, a1, false, b, (short)0, acc1, false, false);
}

// ---------------- kernel 2: one LSTM time step (one direction) ----------------
// Grid: 16 blocks x 256 threads. Block handles one m_group (32 batch rows, staged
// in LDS via async copy) x 8 j_tiles (one per wave). Each wave: 2 M-tiles x 4 gates
// = 8 WMMA accumulators over K=1024.
__global__ __launch_bounds__(256)
void lstm_step(const float* __restrict__ x,              // [16384][16]
               const float* __restrict__ w_ih,           // [4096][14]
               const float* __restrict__ b_ih,           // [4096]
               const float* __restrict__ b_hh,           // [4096]
               const unsigned short* __restrict__ whh,   // [4096][1024] bf16
               const unsigned short* __restrict__ hprev, // [64][1024] bf16
               unsigned short* __restrict__ hout,        // [64][1024] bf16
               float* __restrict__ c,                    // [64][1024] f32
               int s) {
  __shared__ __align__(16) unsigned short hsm[LDS_ROWS * HID];  // 64 KiB

  int tid     = threadIdx.x;
  int m_group = blockIdx.x & 1;                // batch rows [m_group*32, +32)
  int wave    = tid >> 5;
  int lane    = tid & 31;
  int j_tile  = (blockIdx.x >> 1) * 8 + wave;  // 0..63
  int half    = lane >> 4;
  int l16     = lane & 15;
  const int kbaseA = half * 16;  // A layout: lanes 0-15 K0-15, lanes 16-31 K16-31
  const int kbaseB = half * 16;  // B layout: lanes 0-15 K0-15, lanes 16-31 K16-31

  // ---- stage h_prev rows for this m_group into LDS ----
  const unsigned short* gsrc = hprev + (size_t)m_group * LDS_ROWS * HID;
#if __has_builtin(__builtin_amdgcn_global_load_async_to_lds_b128)
  #pragma unroll
  for (int i = 0; i < 16; ++i) {
    int chunk = i * 256 + tid;                 // 4096 x 16B chunks
    __builtin_amdgcn_global_load_async_to_lds_b128(
        (gv4i_p)(gsrc + chunk * 8),
        (lv4i_p)(hsm + chunk * 8),
        0, 0);
  }
  wait_async0();
#else
  for (int i = 0; i < 16; ++i) {
    int chunk = i * 256 + tid;
    ((uint4*)hsm)[chunk] = ((const uint4*)gsrc)[chunk];
  }
#endif
  __syncthreads();

  // ---- WMMA main loop ----
  int jcol = j_tile * 16 + l16;
  const unsigned short* ha0 = hsm + (size_t)(0  + l16) * HID;   // local M-tile 0
  const unsigned short* ha1 = hsm + (size_t)(16 + l16) * HID;   // local M-tile 1
  const unsigned short* bw0 = whh + (size_t)(0 * HID + jcol) * HID;
  const unsigned short* bw1 = whh + (size_t)(1 * HID + jcol) * HID;
  const unsigned short* bw2 = whh + (size_t)(2 * HID + jcol) * HID;
  const unsigned short* bw3 = whh + (size_t)(3 * HID + jcol) * HID;

  v8f a00 = {}, a01 = {}, a10 = {}, a11 = {}, a20 = {}, a21 = {}, a30 = {}, a31 = {};
  for (int kk = 0; kk < HID; kk += 32) {
    v16bf a0, a1;
    #pragma unroll
    for (int j = 0; j < 16; ++j) {
      int ka = kk + kbaseA + j;
      a0[j] = us_as_bf16(ha0[ka]);
      a1[j] = us_as_bf16(ha1[ka]);
    }
    __builtin_prefetch(bw0 + kk + 256, 0, 1);   // hint weight stream ahead
    gate_mma(bw0, kk, kbaseB, a0, a1, a00, a01);
    gate_mma(bw1, kk, kbaseB, a0, a1, a10, a11);
    gate_mma(bw2, kk, kbaseB, a0, a1, a20, a21);
    gate_mma(bw3, kk, kbaseB, a0, a1, a30, a31);
  }

  // ---- fused epilogue: input projection + biases + gate math + state update ----
  int j = jcol;
  float wihr[4][DIN];
  float bs[4];
  #pragma unroll
  for (int g = 0; g < 4; ++g) {
    int n = g * HID + j;
    bs[g] = b_ih[n] + b_hh[n];
    const float* wr = w_ih + (size_t)n * DIN;
    #pragma unroll
    for (int d = 0; d < DIN; ++d) wihr[g][d] = wr[d];
  }
  #pragma unroll
  for (int mt = 0; mt < 2; ++mt) {
    #pragma unroll
    for (int r = 0; r < 8; ++r) {
      int brow = m_group * 32 + mt * 16 + r + half * 8;
      const float* xr = x + (size_t)(s * BATCH + brow) * 16;
      float rec[4];
      rec[0] = mt ? a01[r] : a00[r];
      rec[1] = mt ? a11[r] : a10[r];
      rec[2] = mt ? a21[r] : a20[r];
      rec[3] = mt ? a31[r] : a30[r];
      float gate[4];
      #pragma unroll
      for (int g = 0; g < 4; ++g) {
        float v = rec[g] + bs[g];
        #pragma unroll
        for (int d = 0; d < DIN; ++d) v += xr[d] * wihr[g][d];
        gate[g] = v;
      }
      float i_ = sigm(gate[0]);
      float f_ = sigm(gate[1]);
      float gg = tanhf(gate[2]);
      float o_ = sigm(gate[3]);
      size_t idx = (size_t)brow * HID + j;
      float cn = f_ * c[idx] + i_ * gg;
      c[idx] = cn;
      hout[idx] = f32_to_bf16(o_ * tanhf(cn));
    }
  }
}

// ---------------- kernel 3: FC head, bf16 WMMA, M=16384 K=2048 N=64(pad) ----------------
__global__ __launch_bounds__(256)
void fc_wmma(const unsigned short* __restrict__ hf,   // [16384][1024] (flat row q = s*64+b)
             const unsigned short* __restrict__ hb,
             const unsigned short* __restrict__ fcw,  // [64][2048] bf16 (rows>=50 zero)
             const float* __restrict__ fcb,
             float* __restrict__ tag) {               // [16384][64]
  int wave   = (blockIdx.x * blockDim.x + threadIdx.x) >> 5;
  int lane   = threadIdx.x & 31;
  int n_tile = wave & 3;
  int m_tile = wave >> 2;        // 0..1023
  int half   = lane >> 4;
  int l16    = lane & 15;
  const int kbaseA = half * 16;
  const int kbaseB = half * 16;

  int q_a = m_tile * 16 + l16;
  int ncol = n_tile * 16 + l16;
  const unsigned short* bw = fcw + (size_t)ncol * (2 * HID);

  v8f acc = {};
  for (int kk = 0; kk < 2 * HID; kk += 32) {
    const unsigned short* ar = (kk < HID) ? (hf + (size_t)q_a * HID + kk)
                                          : (hb + (size_t)q_a * HID + (kk - HID));
    __builtin_prefetch(bw + kk + 256, 0, 1);
    v16bf a, b;
    #pragma unroll
    for (int j = 0; j < 16; ++j) {
      a[j] = us_as_bf16(ar[kbaseA + j]);
      b[j] = us_as_bf16(bw[kk + kbaseB + j]);
    }
    acc = __builtin_amdgcn_wmma_f32_16x16x32_bf16(false, a, false, b, (short)0, acc, false, false);
  }
  #pragma unroll
  for (int r = 0; r < 8; ++r) {
    int q = m_tile * 16 + r + half * 8;
    float v = acc[r] + ((ncol < TAGS) ? fcb[ncol] : 0.f);
    tag[(size_t)q * TAGP + ncol] = v;
  }
}

// ---------------- kernel 4: log_softmax over sequence axis (axis=1) ----------------
__global__ void logsoftmax_seq(const float* __restrict__ tag, float* __restrict__ out) {
  int b = blockIdx.x;          // 64
  int t = threadIdx.x;         // 64 threads, 50 active
  if (t >= TAGS) return;
  float mx = -3.4e38f;
  for (int s = 0; s < SEQ; ++s)
    mx = fmaxf(mx, tag[((size_t)b * SEQ + s) * TAGP + t]);
  float sum = 0.f;
  for (int s = 0; s < SEQ; ++s)
    sum += expf(tag[((size_t)b * SEQ + s) * TAGP + t] - mx);
  float lse = logf(sum) + mx;
  for (int s = 0; s < SEQ; ++s)
    out[((size_t)b * SEQ + s) * TAGS + t] =
        tag[((size_t)b * SEQ + s) * TAGP + t] - lse;
}

// ---------------- host orchestration ----------------
extern "C" void kernel_launch(void* const* d_in, const int* in_sizes, int n_in,
                              void* d_out, int out_size, void* d_ws, size_t ws_size,
                              hipStream_t stream) {
  const int*   char_idx = (const int*)  d_in[0];
  const int*   sent     = (const int*)  d_in[1];
  const float* wemb     = (const float*)d_in[2];
  const float* cemb     = (const float*)d_in[3];
  const float* cnnw     = (const float*)d_in[4];
  const float* cnnb     = (const float*)d_in[5];
  const float* wihf     = (const float*)d_in[6];
  const float* whhf     = (const float*)d_in[7];
  const float* bihf     = (const float*)d_in[8];
  const float* bhhf     = (const float*)d_in[9];
  const float* wihb     = (const float*)d_in[10];
  const float* whhb     = (const float*)d_in[11];
  const float* bihb     = (const float*)d_in[12];
  const float* bhhb     = (const float*)d_in[13];
  const float* fcw      = (const float*)d_in[14];
  const float* fcb      = (const float*)d_in[15];

  char* ws = (char*)d_ws;
  float*          x       = (float*)         (ws + OFF_X);
  unsigned short* whhf16  = (unsigned short*)(ws + OFF_WHHF);
  unsigned short* whhb16  = (unsigned short*)(ws + OFF_WHHB);
  unsigned short* hsf     = (unsigned short*)(ws + OFF_HSF);
  unsigned short* hsb     = (unsigned short*)(ws + OFF_HSB);
  float*          c_f     = (float*)         (ws + OFF_CF);
  float*          c_b     = (float*)         (ws + OFF_CB);
  unsigned short* hzero   = (unsigned short*)(ws + OFF_HZERO);
  unsigned short* fcw16   = (unsigned short*)(ws + OFF_FCW);
  float*          tag     = (float*)         (ws + OFF_TAG);

  // 1) embeddings + char CNN
  prep_embed<<<(ROWS + 255) / 256, 256, 0, stream>>>(char_idx, sent, wemb, cemb, cnnw, cnnb, x);

  // 2) weight conversion + state init
  int nwh = GATES * HID;
  cvt_bf16<<<(nwh + 255) / 256, 256, 0, stream>>>(whhf, whhf16, nwh);
  cvt_bf16<<<(nwh + 255) / 256, 256, 0, stream>>>(whhb, whhb16, nwh);
  cvt_fcw<<<(TAGP * 2 * HID + 255) / 256, 256, 0, stream>>>(fcw, fcw16);
  // zero c_f, c_b (f32) and hzero (bf16) — contiguous region starting at OFF_CF
  int nz = (2 * BATCH * HID * 4 + BATCH * HID * 2) / 4;
  zero_u32<<<(nz + 255) / 256, 256, 0, stream>>>((unsigned int*)c_f, nz);

  // 3) bidirectional LSTM: forward step t and backward step (255-t) per iteration
  for (int t = 0; t < SEQ; ++t) {
    const unsigned short* hp_f = (t == 0) ? hzero : (hsf + (size_t)(t - 1) * BATCH * HID);
    lstm_step<<<16, 256, 0, stream>>>(x, wihf, bihf, bhhf, whhf16,
                                      hp_f, hsf + (size_t)t * BATCH * HID, c_f, t);
    int sb = SEQ - 1 - t;
    const unsigned short* hp_b = (t == 0) ? hzero : (hsb + (size_t)(sb + 1) * BATCH * HID);
    lstm_step<<<16, 256, 0, stream>>>(x, wihb, bihb, bhhb, whhb16,
                                      hp_b, hsb + (size_t)sb * BATCH * HID, c_b, sb);
  }

  // 4) FC head (16384x2048 @ 2048x64) via bf16 WMMA
  fc_wmma<<<512, 256, 0, stream>>>(hsf, hsb, fcw16, fcb, tag);

  // 5) log_softmax over sequence axis, write final output
  logsoftmax_seq<<<64, 64, 0, stream>>>(tag, (float*)d_out);
}
